// ScannedRNN_15479062134961
// MI455X (gfx1250) — compile-verified, matrix operand order checked by reading, not run on hardware
//
#include <hip/hip_runtime.h>
#include <hip/hip_bf16.h>

// ---------------- problem constants ----------------
#define TT   1024
#define BB   128
#define DD   512
#define NCOL 3072              // C cols: [x@Wi (1536)] | [h@Whcat (1536)]
#define NBLK 96                // persistent blocks
#define NTHR 256               // 8 wave32 per block
#define NWAVE 8
#define PW_ELEMS (96*16*32*16)           // packed weight buffer elems (bf16)
#define STEP_AELEMS (8*16*32*16)         // packed A elems per step = B*D
#define SLICE 8192                       // packed A elems per row-tile slice (16KB)

typedef __attribute__((ext_vector_type(16))) __bf16 v16bf;
typedef __attribute__((ext_vector_type(8)))  float  v8f;

// A/B fragment element mapping (v_wmma_f32_16x16x32_bf16, wave32):
// lane L: row/col = L%16, H = L/16 ; element e: K = kb*32 + (e<8 ? H*8+e : 16+H*8+(e-8))
// Packed flat index within one [16 x 512] row-tile slice:
//   kb*512 + lane*16 + e   (each lane's 16 bf16 = 32B contiguous); slice = rowt*8192
__device__ __host__ __forceinline__ int afrag_index(int b, int d) {
    int rowt = b >> 4, M = b & 15;
    int kb = d >> 5, ks = d & 31;
    int H = (ks >> 3) & 1;
    int e = (ks & 7) + ((ks >> 4) << 3);
    int lane = M + (H << 4);
    return rowt * SLICE + kb * 512 + lane * 16 + e;
}

// ---------------- grid-wide barrier ----------------
__device__ __forceinline__ void grid_sync(unsigned* counter, unsigned* release,
                                          unsigned idx, unsigned nblk) {
    __syncthreads();
    if (threadIdx.x == 0) {
        __threadfence();   // publish this block's prior writes (agent scope)
        unsigned prev = __hip_atomic_fetch_add(counter, 1u, __ATOMIC_ACQ_REL,
                                               __HIP_MEMORY_SCOPE_AGENT);
        if (prev == idx * nblk + (nblk - 1u)) {
            __hip_atomic_store(release, idx + 1u, __ATOMIC_RELEASE,
                               __HIP_MEMORY_SCOPE_AGENT);
        } else {
            while (__hip_atomic_load(release, __ATOMIC_ACQUIRE,
                                     __HIP_MEMORY_SCOPE_AGENT) < idx + 1u) {
                __builtin_amdgcn_s_sleep(1);
            }
        }
        __threadfence();   // acquire: make other blocks' writes visible
    }
    __syncthreads();
}

// ---------------- weight packing: f32 -> bf16 in WMMA B-fragment order ----
__global__ __launch_bounds__(256)
void pack_weights_kernel(const float* __restrict__ Wi,
                         const float* __restrict__ Whrz,
                         const float* __restrict__ Whn,
                         __bf16* __restrict__ PWX,
                         __bf16* __restrict__ PWH) {
    int i = blockIdx.x * blockDim.x + threadIdx.x;
    if (i >= 2 * PW_ELEMS) return;
    int buf  = i / PW_ELEMS;
    int rem  = i % PW_ELEMS;
    int ct   = rem >> 13;
    int kb   = (rem >> 9) & 15;
    int lane = (rem >> 4) & 31;
    int e    = rem & 15;
    int N = lane & 15, H = lane >> 4;
    int K = kb * 32 + ((e < 8) ? (H * 8 + e) : (16 + H * 8 + (e - 8)));
    if (buf == 0) {
        PWX[rem] = (__bf16)Wi[(size_t)K * 1536 + ct * 16 + N];
    } else {
        float w = (ct < 64) ? Whrz[(size_t)K * 1024 + ct * 16 + N]
                            : Whn[(size_t)K * 512 + (ct - 64) * 16 + N];
        PWH[rem] = (__bf16)w;
    }
}

// ---------------- x packing: ins f32 -> bf16 in WMMA A-fragment order ----
__global__ __launch_bounds__(256)
void pack_x_kernel(const float* __restrict__ ins, __bf16* __restrict__ PAX) {
    size_t i0 = ((size_t)blockIdx.x * blockDim.x + threadIdx.x) * 4;
#pragma unroll
    for (int k = 0; k < 4; ++k) {
        size_t i = i0 + k;
        if (i >= (size_t)TT * BB * DD) return;
        int t = (int)(i >> 16);
        int r = (int)(i & 65535);
        int b = r >> 9, d = r & 511;
        PAX[(size_t)t * STEP_AELEMS + afrag_index(b, d)] = (__bf16)ins[i];
    }
}

// fallback staging helpers: thread tid covers row M=tid>>4 of the slice, kb=tid&15
__device__ __forceinline__ void load_x_f32_regs(const float* __restrict__ xbase,
                                                int rowt, int tid, float4 r[8]) {
    int b  = rowt * 16 + (tid >> 4);
    int d0 = (tid & 15) * 32;
    const float4* p = (const float4*)(xbase + (size_t)b * DD + d0);
#pragma unroll
    for (int i = 0; i < 8; ++i) r[i] = p[i];
}
__device__ __forceinline__ void store_x_lds(__bf16* __restrict__ xbuf, int tid,
                                            const float4 r[8]) {
    int kb = tid & 15, M = tid >> 4;
    // H=0 run: ks {0..7,16..23} = r[0],r[1],r[4],r[5]; H=1 run: ks {8..15,24..31}
    v16bf t0, t1;
    t0[0]=(__bf16)r[0].x; t0[1]=(__bf16)r[0].y; t0[2]=(__bf16)r[0].z; t0[3]=(__bf16)r[0].w;
    t0[4]=(__bf16)r[1].x; t0[5]=(__bf16)r[1].y; t0[6]=(__bf16)r[1].z; t0[7]=(__bf16)r[1].w;
    t0[8]=(__bf16)r[4].x; t0[9]=(__bf16)r[4].y; t0[10]=(__bf16)r[4].z; t0[11]=(__bf16)r[4].w;
    t0[12]=(__bf16)r[5].x; t0[13]=(__bf16)r[5].y; t0[14]=(__bf16)r[5].z; t0[15]=(__bf16)r[5].w;
    t1[0]=(__bf16)r[2].x; t1[1]=(__bf16)r[2].y; t1[2]=(__bf16)r[2].z; t1[3]=(__bf16)r[2].w;
    t1[4]=(__bf16)r[3].x; t1[5]=(__bf16)r[3].y; t1[6]=(__bf16)r[3].z; t1[7]=(__bf16)r[3].w;
    t1[8]=(__bf16)r[6].x; t1[9]=(__bf16)r[6].y; t1[10]=(__bf16)r[6].z; t1[11]=(__bf16)r[6].w;
    t1[12]=(__bf16)r[7].x; t1[13]=(__bf16)r[7].y; t1[14]=(__bf16)r[7].z; t1[15]=(__bf16)r[7].w;
    *(v16bf*)(xbuf + kb * 512 + M * 16)        = t0;   // lane M   (H=0)
    *(v16bf*)(xbuf + kb * 512 + (M + 16) * 16) = t1;   // lane M+16 (H=1)
}

// ---------------- persistent GRU scan kernel ----------------
// Block owns one rowt (j&7); its 8 waves own 8 consecutive ct's. A slices are
// staged into LDS once per step (x double-buffered one step ahead); B fragments
// live in VGPRs for the entire scan.
template <bool PACKED_X>
__global__ __launch_bounds__(NTHR)
void gru_scan_kernel(const float* __restrict__ ins,
                     const __bf16* __restrict__ PAX,
                     const int*   __restrict__ resets,
                     const float* __restrict__ h0,
                     const float* __restrict__ bi,
                     const float* __restrict__ bhn,
                     const __bf16* __restrict__ PWX,
                     const __bf16* __restrict__ PWH,
                     float* __restrict__ hF,            // [2,B,D] f32, reset-applied
                     __bf16* __restrict__ hBF,          // [2,STEP_AELEMS] packed
                     float* __restrict__ cbuf,          // [B,NCOL]
                     float* __restrict__ out,           // [T,B,D]
                     unsigned* __restrict__ sync_counter,
                     unsigned* __restrict__ sync_release) {
    __shared__ __bf16 hs[SLICE];       // 16KB: h slice for this rowt
    __shared__ __bf16 xs[2][SLICE];    // 32KB: x slices, double buffered

    const int j    = blockIdx.x;
    const int tid  = threadIdx.x;
    const int lane = tid & 31;
    const int wave = tid >> 5;
    const int gtid = j * NTHR + tid;
    const int rowt = j & 7;
    const int ct   = (j >> 3) * 8 + wave;   // 0..95
    const int H    = lane >> 4;
    const int N    = lane & 15;
    unsigned bar = 0;

    // --- preload this wave's B fragments into VGPRs (held across all steps) ---
    v16bf bx[16], bh[16];
#pragma unroll
    for (int kb = 0; kb < 16; ++kb) {
        bx[kb] = *(const v16bf*)(PWX + (size_t)ct * SLICE + kb * 512 + lane * 16);
        bh[kb] = *(const v16bf*)(PWH + (size_t)ct * SLICE + kb * 512 + lane * 16);
    }

    // --- init h state (apply resets[0]) ---
    for (int idx = gtid; idx < BB * DD; idx += NBLK * NTHR) {
        int b = idx >> 9, d = idx & 511;
        float v = (resets[b] != 0) ? 0.0f : h0[idx];
        hF[idx] = v;
        hBF[afrag_index(b, d)] = (__bf16)v;
    }
    // --- stage x(0) into xs[0] ---
    if (PACKED_X) {
        const uint4* s = (const uint4*)(PAX + (size_t)rowt * SLICE) + tid * 4;
        uint4 a0 = s[0], a1 = s[1], a2 = s[2], a3 = s[3];
        uint4* d = (uint4*)&xs[0][0] + tid * 4;
        d[0] = a0; d[1] = a1; d[2] = a2; d[3] = a3;
    } else {
        float4 xr[8];
        load_x_f32_regs(ins, rowt, tid, xr);
        store_x_lds(&xs[0][0], tid, xr);
    }
    grid_sync(sync_counter, sync_release, bar++, NBLK);

    for (int t = 0; t < TT; ++t) {
        const float*  hcurF = hF + (t & 1) * (BB * DD);
        float*        hnxtF = hF + ((t + 1) & 1) * (BB * DD);
        const __bf16* hcurB = hBF + (t & 1) * STEP_AELEMS;
        __bf16*       hnxtB = hBF + ((t + 1) & 1) * STEP_AELEMS;
        const bool    pre   = (t + 1 < TT);

        // 1) issue h(t) slice loads (one burst, single L2 latency)
        const uint4* hsrc = (const uint4*)(hcurB + (size_t)rowt * SLICE) + tid * 4;
        uint4 hr0 = hsrc[0], hr1 = hsrc[1], hr2 = hsrc[2], hr3 = hsrc[3];

        // 2) issue x(t+1) prefetch loads (consumed after the WMMA loop)
        uint4 xp0, xp1, xp2, xp3;
        float4 xrp[8];
        if (PACKED_X) {
            if (pre) {
                const uint4* xsrc =
                    (const uint4*)(PAX + (size_t)(t + 1) * STEP_AELEMS + (size_t)rowt * SLICE) + tid * 4;
                xp0 = xsrc[0]; xp1 = xsrc[1]; xp2 = xsrc[2]; xp3 = xsrc[3];
            }
        } else {
            if (pre) load_x_f32_regs(ins + (size_t)(t + 1) * (BB * DD), rowt, tid, xrp);
        }

        // 3) stage h into LDS
        uint4* hdst = (uint4*)hs + tid * 4;
        hdst[0] = hr0; hdst[1] = hr1; hdst[2] = hr2; hdst[3] = hr3;
        __syncthreads();

        // 4) WMMA loop: A from LDS, B from registers
        v8f accx = {}, acch = {};
        const __bf16* xcur = &xs[t & 1][0];
#pragma unroll
        for (int kb = 0; kb < 16; ++kb) {
            v16bf ax = *(const v16bf*)(xcur + kb * 512 + lane * 16);
            v16bf ah = *(const v16bf*)(hs + kb * 512 + lane * 16);
            accx = __builtin_amdgcn_wmma_f32_16x16x32_bf16(
                false, ax, false, bx[kb], (short)0, accx, false, false);
            acch = __builtin_amdgcn_wmma_f32_16x16x32_bf16(
                false, ah, false, bh[kb], (short)0, acch, false, false);
        }

        // 5) commit x(t+1) prefetch into the other LDS buffer
        if (pre) {
            if (PACKED_X) {
                uint4* d = (uint4*)&xs[(t + 1) & 1][0] + tid * 4;
                d[0] = xp0; d[1] = xp1; d[2] = xp2; d[3] = xp3;
            } else {
                store_x_lds(&xs[(t + 1) & 1][0], tid, xrp);
            }
        }

        // store C tiles: lane covers col N, rows M = v + 8*H
        float* cx = cbuf + (size_t)(rowt * 16 + H * 8) * NCOL + ct * 16 + N;
        float* ch = cx + 1536;
#pragma unroll
        for (int v = 0; v < 8; ++v) {
            cx[(size_t)v * NCOL] = accx[v];
            ch[(size_t)v * NCOL] = acch[v];
        }
        grid_sync(sync_counter, sync_release, bar++, NBLK);

        // ---- phase 2: gates + update + clip; write ys[t] and next h ----
        for (int idx = gtid; idx < BB * DD; idx += NBLK * NTHR) {
            int b = idx >> 9, d = idx & 511;
            const float* crow = cbuf + (size_t)b * NCOL;
            float pr = crow[d]          + crow[1536 + d]       + bi[d];
            float pz = crow[DD + d]     + crow[1536 + DD + d]  + bi[DD + d];
            float nx = crow[2 * DD + d] + bi[2 * DD + d];
            float nh = crow[2560 + d]   + bhn[d];
            float r  = 1.0f / (1.0f + __expf(-pr));
            float z  = 1.0f / (1.0f + __expf(-pz));
            float pn = nx + r * nh;
            float e2 = __expf(-2.0f * pn);
            float n  = (1.0f - e2) / (1.0f + e2);
            float hp = hcurF[idx];                 // reset already applied
            float hn = (1.0f - z) * n + z * hp;
            hn = fminf(1.0f, fmaxf(-1.0f, hn));
            out[(size_t)t * (BB * DD) + idx] = hn;
            float hv = hn;
            if (t + 1 < TT && resets[(t + 1) * BB + b] != 0) hv = 0.0f;
            hnxtF[idx] = hv;
            hnxtB[afrag_index(b, d)] = (__bf16)hv;
        }
        grid_sync(sync_counter, sync_release, bar++, NBLK);
    }
}

// ---------------- host launch ----------------
extern "C" void kernel_launch(void* const* d_in, const int* in_sizes, int n_in,
                              void* d_out, int out_size, void* d_ws, size_t ws_size,
                              hipStream_t stream) {
    (void)in_sizes; (void)n_in; (void)out_size;
    const float* h0   = (const float*)d_in[0];
    const float* ins  = (const float*)d_in[1];
    const int*   rst  = (const int*)d_in[2];
    const float* Wi   = (const float*)d_in[3];
    const float* bi   = (const float*)d_in[4];
    const float* Whrz = (const float*)d_in[5];
    const float* Whn  = (const float*)d_in[6];
    const float* bhn  = (const float*)d_in[7];
    float* out = (float*)d_out;

    char* ws = (char*)d_ws;
    unsigned* syncc = (unsigned*)ws;
    unsigned* syncr = (unsigned*)(ws + 64);
    __bf16* PWX  = (__bf16*)(ws + 256);
    __bf16* PWH  = PWX + PW_ELEMS;
    float*  hF   = (float*)((char*)(PWH + PW_ELEMS));
    __bf16* hBF  = (__bf16*)(hF + 2 * BB * DD);
    float*  cbuf = (float*)(hBF + 2 * STEP_AELEMS);
    __bf16* PAX  = (__bf16*)(cbuf + BB * NCOL);

    size_t need_base = (size_t)((char*)PAX - ws);                 // ~5.5 MB
    size_t need_pax  = need_base + (size_t)TT * STEP_AELEMS * 2;  // +128 MB
    bool packed = (ws_size >= need_pax);

    hipMemsetAsync(ws, 0, 256, stream);

    pack_weights_kernel<<<dim3((2 * PW_ELEMS + 255) / 256), dim3(256), 0, stream>>>(
        Wi, Whrz, Whn, PWX, PWH);

    if (packed) {
        size_t tot = (size_t)TT * BB * DD;
        int blocks = (int)((tot / 4 + 255) / 256);
        pack_x_kernel<<<dim3(blocks), dim3(256), 0, stream>>>(ins, PAX);
        gru_scan_kernel<true><<<dim3(NBLK), dim3(NTHR), 0, stream>>>(
            ins, PAX, rst, h0, bi, bhn, PWX, PWH, hF, hBF, cbuf, out, syncc, syncr);
    } else {
        gru_scan_kernel<false><<<dim3(NBLK), dim3(NTHR), 0, stream>>>(
            ins, PAX, rst, h0, bi, bhn, PWX, PWH, hF, hBF, cbuf, out, syncc, syncr);
    }
}
